// TransformerAttention_90924457656878
// MI455X (gfx1250) — matrix operand order repaired and checked by reading, not compiled, not run on hardware
//
#include <hip/hip_runtime.h>

typedef __attribute__((ext_vector_type(16))) _Float16 v16h;
typedef __attribute__((ext_vector_type(8)))  float    v8f;

#define B_  2
#define T_  2048
#define H_  16
#define D_  64
#define M_  1024
#define N_  1024   /* H*D */

// ---------------- WMMA fragment helpers (CDNA5 16x16x32 f16, wave32) ----------
// All loaders are two aligned 16B LDS reads (ds_load_b128 x2).
// Requires: ld multiple of 8 halves, k0 multiple of 8.

// A (16x32 rows, row-major): lane l=lane&15 -> row; hi=lane>>4 selects K-half.
// a[0..7] = row[k0 + hi*8 + 0..7], a[8..15] = row[k0 + 16 + hi*8 + 0..7].
static __device__ __forceinline__ v16h load_a_frag(const _Float16* lds, int ld,
                                                   int row0, int k0, int lane) {
  const int l = lane & 15, hi = (lane >> 4) & 1;
  const _Float16* p = lds + (row0 + l) * ld + k0 + (hi << 3);
  union { uint4 u[2]; v16h h; } r;
  r.u[0] = *(const uint4*)(p);
  r.u[1] = *(const uint4*)(p + 16);
  return r.h;
}

// B (32xN) with element (k,n) stored at lds[(n0+n)*ld + k0+k] (N-major,
// K contiguous). b[0..7] = K hi*8+0..7, b[8..15] = K 16+hi*8+0..7 (matches A).
static __device__ __forceinline__ v16h load_bT_frag(const _Float16* lds, int ld,
                                                    int n0, int k0, int lane) {
  const int l = lane & 15, hi = (lane >> 4) & 1;
  const _Float16* p = lds + (n0 + l) * ld + k0 + (hi << 3);
  union { uint4 u[2]; v16h h; } r;
  r.u[0] = *(const uint4*)(p);
  r.u[1] = *(const uint4*)(p + 16);
  return r.h;
}

static __device__ __forceinline__ v8f zero8() {
  v8f z = {0.f, 0.f, 0.f, 0.f, 0.f, 0.f, 0.f, 0.f};
  return z;
}

static __device__ __forceinline__ unsigned pack2f(float x, float y) {
  union { _Float16 h[2]; unsigned u; } r;
  r.h[0] = (_Float16)x;
  r.h[1] = (_Float16)y;
  return r.u;
}

static __device__ __forceinline__ unsigned pack2h(_Float16 x, _Float16 y) {
  union { _Float16 h[2]; unsigned u; } r;
  r.h[0] = x;
  r.h[1] = y;
  return r.u;
}

// ---------------- Kernel 1: QKV projection GEMM (X[4096,1024] @ W[1024,1024]) --
// grid (N/128, rows/128, 3), block 256 (8 waves). Output f16 [B,H,T,D] in ws.
__global__ void __launch_bounds__(256)
qkv_gemm_kernel(const float* __restrict__ X,
                const float* __restrict__ Wq, const float* __restrict__ bq,
                const float* __restrict__ Wk, const float* __restrict__ bk,
                const float* __restrict__ Wv, const float* __restrict__ bv,
                _Float16* __restrict__ qf, _Float16* __restrict__ kf,
                _Float16* __restrict__ vf) {
  __shared__ _Float16 lx[128][40];    // 128x32 A tile, row-major (+pad)
  __shared__ _Float16 lwT[128][40];   // 32x128 B tile, stored N-major: lwT[n][k]

  const int tid  = threadIdx.x;
  const int lane = tid & 31, wid = tid >> 5;
  const int wr = wid >> 1, wc = wid & 1;           // 4x2 wave grid -> 32x64 each
  const int r0 = blockIdx.y * 128;
  const int c0 = blockIdx.x * 128;
  const int z  = blockIdx.z;

  const float* W    = (z == 0) ? Wq : (z == 1) ? Wk : Wv;
  const float* bias = (z == 0) ? bq : (z == 1) ? bk : bv;
  _Float16*    dst  = (z == 0) ? qf : (z == 1) ? kf : vf;

  v8f acc[2][4];
#pragma unroll
  for (int i = 0; i < 2; ++i)
#pragma unroll
    for (int j = 0; j < 4; ++j) acc[i][j] = zero8();

  const int xrow = tid >> 1, xc = (tid & 1) << 4;   // X: 16 floats each
  const int wkp  = tid >> 4;                        // W: k-pair 0..15
  const int wns  = (tid & 15) << 3;                 // W: 8 n-values

  for (int kk = 0; kk < M_; kk += 32) {
    // --- stage X tile (packed u32 pair stores) ---
    const float4* xs = (const float4*)(X + (size_t)(r0 + xrow) * M_ + kk + xc);
#pragma unroll
    for (int q = 0; q < 4; ++q) {
      float4 t4 = xs[q];
      uint2 pk;
      pk.x = pack2f(t4.x, t4.y);
      pk.y = pack2f(t4.z, t4.w);
      *(uint2*)(&lx[xrow][xc + 4 * q]) = pk;
    }
    // --- stage W tile transposed: lwT[n][k], k-pairs packed per u32 ---
    {
      const float* wr0 = W + (size_t)(kk + 2 * wkp) * N_ + c0 + wns;
      const float* wr1 = wr0 + N_;
      float4 a0 = *(const float4*)(wr0);
      float4 a1 = *(const float4*)(wr0 + 4);
      float4 b0 = *(const float4*)(wr1);
      float4 b1 = *(const float4*)(wr1 + 4);
      *(unsigned*)(&lwT[wns + 0][2 * wkp]) = pack2f(a0.x, b0.x);
      *(unsigned*)(&lwT[wns + 1][2 * wkp]) = pack2f(a0.y, b0.y);
      *(unsigned*)(&lwT[wns + 2][2 * wkp]) = pack2f(a0.z, b0.z);
      *(unsigned*)(&lwT[wns + 3][2 * wkp]) = pack2f(a0.w, b0.w);
      *(unsigned*)(&lwT[wns + 4][2 * wkp]) = pack2f(a1.x, b1.x);
      *(unsigned*)(&lwT[wns + 5][2 * wkp]) = pack2f(a1.y, b1.y);
      *(unsigned*)(&lwT[wns + 6][2 * wkp]) = pack2f(a1.z, b1.z);
      *(unsigned*)(&lwT[wns + 7][2 * wkp]) = pack2f(a1.w, b1.w);
    }
    if (kk + 32 < M_) {  // global_prefetch_b8 of next tiles
      __builtin_prefetch(X + (size_t)(r0 + xrow) * M_ + kk + 32 + xc, 0, 1);
      __builtin_prefetch(W + (size_t)(kk + 32 + 2 * wkp) * N_ + c0 + wns, 0, 1);
    }
    __syncthreads();

    v16h afr[2];
#pragma unroll
    for (int rt = 0; rt < 2; ++rt)
      afr[rt] = load_a_frag(&lx[0][0], 40, wr * 32 + rt * 16, 0, lane);
#pragma unroll
    for (int ct = 0; ct < 4; ++ct) {
      v16h bfr = load_bT_frag(&lwT[0][0], 40, wc * 64 + ct * 16, 0, lane);
#pragma unroll
      for (int rt = 0; rt < 2; ++rt)
        acc[rt][ct] = __builtin_amdgcn_wmma_f32_16x16x32_f16(
            false, afr[rt], false, bfr, (short)0, acc[rt][ct], false, false);
    }
    __syncthreads();
  }

  const int l = lane & 15, hi = (lane >> 4) & 1;
#pragma unroll
  for (int rt = 0; rt < 2; ++rt) {
#pragma unroll
    for (int ct = 0; ct < 4; ++ct) {
      const int gc = c0 + wc * 64 + ct * 16 + l;
      const int h = gc >> 6, d = gc & 63;
      const float bb = bias[gc];
#pragma unroll
      for (int r = 0; r < 8; ++r) {
        const int gr = r0 + wr * 32 + rt * 16 + hi * 8 + r;   // row in [0,4096)
        const int b = gr >> 11, t = gr & 2047;
        dst[(((size_t)(b * H_ + h) * T_ + t) << 6) + d] =
            (_Float16)(acc[rt][ct][r] + bb);
      }
    }
  }
}

// ---------------- Kernel 2: scores = Q K^T / 8 + mask -------------------------
// grid (T/128, T/128, B*H), block 256. D=64 fits in one LDS pass.
__global__ void __launch_bounds__(256)
scores_kernel(const _Float16* __restrict__ qf, const _Float16* __restrict__ kf,
              const float* __restrict__ mask, float* __restrict__ scores) {
  __shared__ _Float16 lq[128][72];
  __shared__ _Float16 lk[128][72];

  const int tid = threadIdx.x, lane = tid & 31, wid = tid >> 5;
  const int wr = wid >> 1, wc = wid & 1;
  const int bh = blockIdx.z, b = bh >> 4;
  const int q0 = blockIdx.y * 128, k0 = blockIdx.x * 128;

  const int row = tid >> 1, c0 = (tid & 1) << 5;  // 32 f16 = 64B per thread
  {
    const uint4* sq = (const uint4*)(qf + (((size_t)bh * T_ + q0 + row) << 6) + c0);
    uint4* dq = (uint4*)(&lq[row][c0]);
#pragma unroll
    for (int q = 0; q < 4; ++q) dq[q] = sq[q];
    const uint4* sk = (const uint4*)(kf + (((size_t)bh * T_ + k0 + row) << 6) + c0);
    uint4* dk = (uint4*)(&lk[row][c0]);
#pragma unroll
    for (int q = 0; q < 4; ++q) dk[q] = sk[q];
  }
  __syncthreads();

  v8f acc[2][4];
#pragma unroll
  for (int i = 0; i < 2; ++i)
#pragma unroll
    for (int j = 0; j < 4; ++j) acc[i][j] = zero8();

#pragma unroll
  for (int kc = 0; kc < 2; ++kc) {   // D=64 -> two K=32 steps
    v16h afr[2];
#pragma unroll
    for (int rt = 0; rt < 2; ++rt)
      afr[rt] = load_a_frag(&lq[0][0], 72, wr * 32 + rt * 16, kc * 32, lane);
#pragma unroll
    for (int ct = 0; ct < 4; ++ct) {
      v16h bfr = load_bT_frag(&lk[0][0], 72, wc * 64 + ct * 16, kc * 32, lane);
#pragma unroll
      for (int rt = 0; rt < 2; ++rt)
        acc[rt][ct] = __builtin_amdgcn_wmma_f32_16x16x32_f16(
            false, afr[rt], false, bfr, (short)0, acc[rt][ct], false, false);
    }
  }

  const int l = lane & 15, hi = (lane >> 4) & 1;
#pragma unroll
  for (int ct = 0; ct < 4; ++ct) {
    const int gk = k0 + wc * 64 + ct * 16 + l;
    const float madd = (1.0f - mask[b * T_ + gk]) * -10000.0f;
#pragma unroll
    for (int rt = 0; rt < 2; ++rt) {
#pragma unroll
      for (int r = 0; r < 8; ++r) {
        const int gq = q0 + wr * 32 + rt * 16 + hi * 8 + r;
        scores[((size_t)bh * T_ + gq) * T_ + gk] =
            acc[rt][ct][r] * 0.125f + madd;
      }
    }
  }
}

// ---------------- Kernel 3: in-place row softmax over T=2048 ------------------
// grid (B*H*T), block 256 (8 waves), 8 elems/lane.
__global__ void __launch_bounds__(256)
softmax_kernel(float* __restrict__ probs) {
  __shared__ float red[8];
  float* p = probs + (size_t)blockIdx.x * T_;
  const int tid = threadIdx.x;

  float x[8];
  float m = -1e30f;
#pragma unroll
  for (int i = 0; i < 8; ++i) {
    x[i] = p[tid + (i << 8)];
    m = fmaxf(m, x[i]);
  }
#pragma unroll
  for (int off = 16; off > 0; off >>= 1) m = fmaxf(m, __shfl_xor(m, off, 32));
  if ((tid & 31) == 0) red[tid >> 5] = m;
  __syncthreads();
  m = red[0];
#pragma unroll
  for (int w = 1; w < 8; ++w) m = fmaxf(m, red[w]);
  __syncthreads();

  float s = 0.f;
#pragma unroll
  for (int i = 0; i < 8; ++i) {
    x[i] = __expf(x[i] - m);
    s += x[i];
  }
#pragma unroll
  for (int off = 16; off > 0; off >>= 1) s += __shfl_xor(s, off, 32);
  if ((tid & 31) == 0) red[tid >> 5] = s;
  __syncthreads();
  s = red[0];
#pragma unroll
  for (int w = 1; w < 8; ++w) s += red[w];

  const float inv = 1.0f / s;
#pragma unroll
  for (int i = 0; i < 8; ++i) p[tid + (i << 8)] = x[i] * inv;
}

// ---------------- Kernel 4: out = P @ V --------------------------------------
// grid (T/128, B*H), block 256 (8 waves); each wave: 16 rows x 64 cols.
__global__ void __launch_bounds__(256)
pv_kernel(const float* __restrict__ probs, const _Float16* __restrict__ vf,
          float* __restrict__ out) {
  __shared__ _Float16 lp[128][40];   // 128x32 P tile (f16, row-major)
  __shared__ _Float16 lvT[64][40];   // 32x64 V tile stored N-major: lvT[d][t]

  const int tid = threadIdx.x, lane = tid & 31, wid = tid >> 5;
  const int bh = blockIdx.y, b = bh >> 4, h = bh & 15;
  const int q0 = blockIdx.x * 128;

  v8f acc[4];
#pragma unroll
  for (int j = 0; j < 4; ++j) acc[j] = zero8();

  const int prow = tid >> 1, pc = (tid & 1) << 4;  // P: 16 floats each
  const int vtp  = tid >> 4;                       // V: t-pair 0..15
  const int vds  = (tid & 15) << 2;                // V: 4 d-values

  for (int kk = 0; kk < T_; kk += 32) {
    // --- stage P tile (fp32 -> f16, packed u32 pair stores) ---
    const float4* ps =
        (const float4*)(probs + ((size_t)bh * T_ + q0 + prow) * T_ + kk + pc);
#pragma unroll
    for (int q = 0; q < 4; ++q) {
      float4 t4 = ps[q];
      uint2 pk;
      pk.x = pack2f(t4.x, t4.y);
      pk.y = pack2f(t4.z, t4.w);
      *(uint2*)(&lp[prow][pc + 4 * q]) = pk;
    }
    // --- stage V tile transposed: lvT[d][t], t-pairs packed per u32 ---
    {
      const _Float16* v0 = vf + (((size_t)bh * T_ + kk + 2 * vtp) << 6) + vds;
      const _Float16* v1 = v0 + D_;
      union { uint2 v; _Float16 h[4]; } r0, r1;
      r0.v = *(const uint2*)(v0);
      r1.v = *(const uint2*)(v1);
#pragma unroll
      for (int j = 0; j < 4; ++j)
        *(unsigned*)(&lvT[vds + j][2 * vtp]) = pack2h(r0.h[j], r1.h[j]);
    }
    __syncthreads();

    v16h afr = load_a_frag(&lp[0][0], 40, wid * 16, 0, lane);
#pragma unroll
    for (int ct = 0; ct < 4; ++ct) {
      v16h bfr = load_bT_frag(&lvT[0][0], 40, ct * 16, 0, lane);
      acc[ct] = __builtin_amdgcn_wmma_f32_16x16x32_f16(
          false, afr, false, bfr, (short)0, acc[ct], false, false);
    }
    __syncthreads();
  }

  const int l = lane & 15, hi = (lane >> 4) & 1;
#pragma unroll
  for (int ct = 0; ct < 4; ++ct) {
#pragma unroll
    for (int r = 0; r < 8; ++r) {
      const int gt = q0 + wid * 16 + hi * 8 + r;
      out[((size_t)(b * T_ + gt) << 10) + (h << 6) + ct * 16 + l] = acc[ct][r];
    }
  }
}

// ---------------- launcher ----------------------------------------------------
extern "C" void kernel_launch(void* const* d_in, const int* in_sizes, int n_in,
                              void* d_out, int out_size, void* d_ws,
                              size_t ws_size, hipStream_t stream) {
  const float* x    = (const float*)d_in[0];
  const float* mask = (const float*)d_in[1];
  const float* Wq   = (const float*)d_in[2];
  const float* bq   = (const float*)d_in[3];
  const float* Wk   = (const float*)d_in[4];
  const float* bk   = (const float*)d_in[5];
  const float* Wv   = (const float*)d_in[6];
  const float* bv   = (const float*)d_in[7];

  float* out   = (float*)d_out;                     // [B,T,H*D]
  float* probs = out + (size_t)B_ * T_ * H_ * D_;   // [B,H,T,T]

  // Workspace: Q,K,V in f16 [B,H,T,D] -> 3 * 8 MB = 24 MB.
  _Float16* qf = (_Float16*)d_ws;
  _Float16* kf = qf + (size_t)B_ * H_ * T_ * D_;
  _Float16* vf = kf + (size_t)B_ * H_ * T_ * D_;

  qkv_gemm_kernel<<<dim3(N_ / 128, (B_ * T_) / 128, 3), 256, 0, stream>>>(
      x, Wq, bq, Wk, bk, Wv, bv, qf, kf, vf);
  scores_kernel<<<dim3(T_ / 128, T_ / 128, B_ * H_), 256, 0, stream>>>(
      qf, kf, mask, probs);
  softmax_kernel<<<dim3(B_ * H_ * T_), 256, 0, stream>>>(probs);
  pv_kernel<<<dim3(T_ / 128, B_ * H_), 256, 0, stream>>>(probs, vf, out);
}
